// TriangleAttention_19112604467461
// MI455X (gfx1250) — matrix operand order repaired and checked by reading, not compiled, not run on hardware
//
#include <hip/hip_runtime.h>
#include <hip/hip_bf16.h>
#include <math.h>

#define N_RES 384
#define C_Z   128
#define N_HEAD 4
#define NN (N_RES * N_RES)

typedef __attribute__((ext_vector_type(16))) __bf16 v16bf;
typedef __attribute__((ext_vector_type(8)))  __bf16 bf16x8;
typedef __attribute__((ext_vector_type(8)))  float  v8f;
typedef __attribute__((ext_vector_type(4)))  __bf16 bf16x4;

// Build a v16bf fragment from two contiguous 16-byte (8 x bf16) runs.
__device__ __forceinline__ v16bf frag2(const __bf16* p0, const __bf16* p1) {
    bf16x8 lo = *(const bf16x8*)p0;
    bf16x8 hi = *(const bf16x8*)p1;
    return __builtin_shufflevector(lo, hi, 0, 1, 2, 3, 4, 5, 6, 7,
                                          8, 9, 10, 11, 12, 13, 14, 15);
}
// A-matrix fragment (16-bit 16x32 layout, ISA 7.12.2): K runs [off,off+8) and [16+off,24+off)
__device__ __forceinline__ v16bf fragA(const __bf16* row, int lane) {
    int off = (lane & 16) ? 8 : 0;
    return frag2(row + off, row + 16 + off);
}
// B-matrix fragment (16-bit 32x16): K runs [off,off+8) and [16+off,24+off), needs K-contiguous storage
__device__ __forceinline__ v16bf fragB(const __bf16* colrun, int lane) {
    int off = (lane & 16) ? 8 : 0;
    return frag2(colrun + off, colrun + 16 + off);
}
// A fragment sourced from an f32 LDS row (probabilities), packed to bf16
__device__ __forceinline__ v16bf fragA_lds(const float* rowp, int lane) {
    int off = (lane & 16) ? 8 : 0;
    float4 a0 = *(const float4*)(rowp + off);
    float4 a1 = *(const float4*)(rowp + off + 4);
    float4 a2 = *(const float4*)(rowp + off + 16);
    float4 a3 = *(const float4*)(rowp + off + 20);
    v16bf r;
    r[0] = (__bf16)a0.x; r[1] = (__bf16)a0.y; r[2]  = (__bf16)a0.z; r[3]  = (__bf16)a0.w;
    r[4] = (__bf16)a1.x; r[5] = (__bf16)a1.y; r[6]  = (__bf16)a1.z; r[7]  = (__bf16)a1.w;
    r[8] = (__bf16)a2.x; r[9] = (__bf16)a2.y; r[10] = (__bf16)a2.z; r[11] = (__bf16)a2.w;
    r[12] = (__bf16)a3.x; r[13] = (__bf16)a3.y; r[14] = (__bf16)a3.z; r[15] = (__bf16)a3.w;
    return r;
}
__device__ __forceinline__ float wsum(float v) {
#pragma unroll
    for (int m = 16; m >= 1; m >>= 1) v += __shfl_xor(v, m, 32);
    return v;
}
__device__ __forceinline__ float wmaxr(float v) {
#pragma unroll
    for (int m = 16; m >= 1; m >>= 1) v = fmaxf(v, __shfl_xor(v, m, 32));
    return v;
}

// ---------------- kernel 1: pack weights, TRANSPOSED so B fragments are K-contiguous ----
// wcatT[col(512)][d(128)], woT[d(128)][hc(128)]
__global__ void ta_pack_weights(const float* __restrict__ Wq, const float* __restrict__ Wk,
                                const float* __restrict__ Wv, const float* __restrict__ Wg,
                                const float* __restrict__ Wo,
                                __bf16* __restrict__ wcatT, __bf16* __restrict__ woT) {
    int idx = blockIdx.x * blockDim.x + threadIdx.x;
    if (idx < C_Z * 512) {
        int col = idx >> 7;          // 0..511  (seg*128 + hc)
        int d   = idx & 127;
        int seg = col >> 7;
        int hc  = col & 127;
        const float* W = (seg == 0) ? Wq : (seg == 1) ? Wk : (seg == 2) ? Wv : Wg;
        wcatT[idx] = (__bf16)W[d * 128 + hc];
    } else {
        int j = idx - C_Z * 512;
        if (j < C_Z * C_Z) {
            int d  = j >> 7;
            int hc = j & 127;
            woT[j] = (__bf16)Wo[hc * 128 + d];   // Wo ref layout [h][c][d]
        }
    }
}

// ---------------- kernel 2: LayerNorm (one wave32 per row) + pair bias ----------------
__global__ void ta_ln_bias(const float* __restrict__ z,
                           const float* __restrict__ gamma,
                           const float* __restrict__ beta,
                           const float* __restrict__ Wb,
                           __bf16* __restrict__ zn,
                           float* __restrict__ biasbuf) {
    int wave = threadIdx.x >> 5;
    int lane = threadIdx.x & 31;
    long row = (long)blockIdx.x * 8 + wave;
    const float4* zr = (const float4*)(z + row * C_Z);
    float4 x = zr[lane];
    float mu = wsum(x.x + x.y + x.z + x.w) * (1.0f / C_Z);
    float d0 = x.x - mu, d1 = x.y - mu, d2 = x.z - mu, d3 = x.w - mu;
    float var = wsum(d0 * d0 + d1 * d1 + d2 * d2 + d3 * d3) * (1.0f / C_Z);
    float inv = rsqrtf(var + 1e-5f);
    float4 gg = ((const float4*)gamma)[lane];
    float4 bb = ((const float4*)beta)[lane];
    float z0 = d0 * inv * gg.x + bb.x;
    float z1 = d1 * inv * gg.y + bb.y;
    float z2 = d2 * inv * gg.z + bb.z;
    float z3 = d3 * inv * gg.w + bb.w;
    bf16x4 o;
    o[0] = (__bf16)z0; o[1] = (__bf16)z1; o[2] = (__bf16)z2; o[3] = (__bf16)z3;
    *(bf16x4*)(zn + row * C_Z + lane * 4) = o;
    int d = lane * 4;
#pragma unroll
    for (int h = 0; h < N_HEAD; ++h) {
        float p = z0 * Wb[(d + 0) * N_HEAD + h] + z1 * Wb[(d + 1) * N_HEAD + h]
                + z2 * Wb[(d + 2) * N_HEAD + h] + z3 * Wb[(d + 3) * N_HEAD + h];
        p = wsum(p);
        if (lane == 0) biasbuf[(long)h * NN + row] = p;
    }
}

// ------- kernel 3: fused q/k/v/g projections, A fragments loaded once and reused -------
// V stored TRANSPOSED (vt[i][hc][j]); gate preact has bg folded in.
__global__ void ta_proj(const __bf16* __restrict__ zn, const __bf16* __restrict__ wcatT,
                        const float* __restrict__ bg,
                        __bf16* __restrict__ qb, __bf16* __restrict__ kb,
                        __bf16* __restrict__ vt, float* __restrict__ gf) {
    int wave = threadIdx.x >> 5;
    int lane = threadIdx.x & 31;
    int tile = blockIdx.x;
    int Nl = lane & 15;
    const __bf16* arow = zn + ((long)tile * 16 + Nl) * C_Z;
    v16bf a0 = fragA(arow +  0, lane);
    v16bf a1 = fragA(arow + 32, lane);
    v16bf a2 = fragA(arow + 64, lane);
    v16bf a3 = fragA(arow + 96, lane);

    int mbase = (lane & 16) ? 8 : 0;
    int col = wave * 16 + Nl;              // hc in 0..127
    int i_idx = tile / (N_RES / 16);       // tile never crosses an i boundary
    int j0    = (tile % (N_RES / 16)) * 16;
    float bgv = bg[col];

#pragma unroll
    for (int seg = 0; seg < 4; ++seg) {    // 0=q 1=k 2=v 3=g (uniform)
        const __bf16* brun = wcatT + (long)(seg * 128 + col) * C_Z;
        v8f acc = {};
        acc = __builtin_amdgcn_wmma_f32_16x16x32_bf16(false, a0, false, fragB(brun +  0, lane),
                                                      (short)0, acc, false, false);
        acc = __builtin_amdgcn_wmma_f32_16x16x32_bf16(false, a1, false, fragB(brun + 32, lane),
                                                      (short)0, acc, false, false);
        acc = __builtin_amdgcn_wmma_f32_16x16x32_bf16(false, a2, false, fragB(brun + 64, lane),
                                                      (short)0, acc, false, false);
        acc = __builtin_amdgcn_wmma_f32_16x16x32_bf16(false, a3, false, fragB(brun + 96, lane),
                                                      (short)0, acc, false, false);
#pragma unroll
        for (int r = 0; r < 8; ++r) {
            float v = acc[r];
            if (seg == 2) {
                vt[((long)i_idx * C_Z + col) * N_RES + j0 + mbase + r] = (__bf16)v;
            } else {
                long off = ((long)tile * 16 + mbase + r) * C_Z + col;
                if      (seg == 0) qb[off] = (__bf16)v;
                else if (seg == 1) kb[off] = (__bf16)v;
                else               gf[off] = v + bgv;
            }
        }
    }
}

// ---------------- kernel 4: attention per (i, h, 16-query tile) ----------------
__global__ void ta_attn(const __bf16* __restrict__ qb, const __bf16* __restrict__ kb,
                        const __bf16* __restrict__ vt, const float* __restrict__ gf,
                        const float* __restrict__ biasbuf,
                        __bf16* __restrict__ go) {
    __shared__ float p[16][N_RES + 16];    // row stride 400 floats (1600 B, 16B-aligned)
    __shared__ float opart[8][16][16];
    int wave = threadIdx.x >> 5;
    int lane = threadIdx.x & 31;
    int jt = blockIdx.x, h = blockIdx.y, i = blockIdx.z;
    const float scale = 0.17677669529663687f;   // 1/sqrt(32)
    long base = (long)i * N_RES;
    int Nl = lane & 15;
    int mbase = (lane & 16) ? 8 : 0;

    // ---- phase 1: logits = scale * Q K^T + bias ----
    v16bf aq = fragA(qb + (base + jt * 16 + Nl) * C_Z + h * 32, lane);
    for (int kt = wave; kt < N_RES / 16; kt += 8) {     // wave-uniform
        v16bf bk = fragB(kb + (base + kt * 16 + Nl) * C_Z + h * 32, lane);
        v8f acc = {};
        acc = __builtin_amdgcn_wmma_f32_16x16x32_bf16(false, aq, false, bk, (short)0, acc,
                                                      false, false);
        const float* bi = biasbuf + (long)h * NN + (long)(jt * 16) * N_RES + kt * 16;
#pragma unroll
        for (int r = 0; r < 8; ++r) {
            int M = mbase + r;
            p[M][kt * 16 + Nl] = acc[r] * scale + bi[(long)M * N_RES + Nl];
        }
    }
    __syncthreads();

    // ---- phase 2: softmax over 384 keys, float4 LDS traffic (2 rows per wave) ----
#pragma unroll
    for (int rr = 0; rr < 2; ++rr) {
        int rowm = wave * 2 + rr;
        float4* rp = (float4*)&p[rowm][0];
        float m = -3.0e38f;
#pragma unroll
        for (int t = 0; t < 3; ++t) {
            float4 x = rp[lane + t * 32];
            m = fmaxf(m, fmaxf(fmaxf(x.x, x.y), fmaxf(x.z, x.w)));
        }
        m = wmaxr(m);
        float s = 0.f;
#pragma unroll
        for (int t = 0; t < 3; ++t) {
            float4 x = rp[lane + t * 32];
            x.x = __expf(x.x - m); x.y = __expf(x.y - m);
            x.z = __expf(x.z - m); x.w = __expf(x.w - m);
            s += x.x + x.y + x.z + x.w;
            rp[lane + t * 32] = x;
        }
        s = wsum(s);
        float invs = 1.0f / s;
#pragma unroll
        for (int t = 0; t < 3; ++t) {
            float4 x = rp[lane + t * 32];
            x.x *= invs; x.y *= invs; x.z *= invs; x.w *= invs;
            rp[lane + t * 32] = x;
        }
    }
    __syncthreads();

    // ---- phase 3: O = P @ V ; V is transposed so B fragments are contiguous ----
    int chalf = wave & 1;
    int kset  = wave >> 1;
    const __bf16* vrun = vt + ((long)i * C_Z + h * 32 + chalf * 16 + Nl) * N_RES;
    v8f acc = {};
    for (int kc = kset; kc < 12; kc += 4) {             // wave-uniform
        int kbase = kc * 32;
        v16bf a = fragA_lds(&p[Nl][kbase], lane);
        v16bf b = fragB(vrun + kbase, lane);
        acc = __builtin_amdgcn_wmma_f32_16x16x32_bf16(false, a, false, b, (short)0, acc,
                                                      false, false);
    }
#pragma unroll
    for (int r = 0; r < 8; ++r) opart[wave][mbase + r][Nl] = acc[r];
    __syncthreads();

    // reduce 4 K-subsets per c-half, sigmoid gating (bg already folded), store bf16
    for (int idx = threadIdx.x; idx < 512; idx += 256) {
        int ch  = idx >> 8;
        int rem = idx & 255;
        int M   = rem >> 4;
        int c16 = rem & 15;
        float s = opart[ch][M][c16] + opart[ch + 2][M][c16]
                + opart[ch + 4][M][c16] + opart[ch + 6][M][c16];
        int c = h * 32 + ch * 16 + c16;
        long off = (base + jt * 16 + M) * C_Z + c;
        float gate = 1.0f / (1.0f + __expf(-gf[off]));
        go[off] = (__bf16)(gate * s);
    }
}

// ---------------- kernel 5: output projection + bo ----------------
__global__ void ta_outproj(const __bf16* __restrict__ go, const __bf16* __restrict__ woT,
                           const float* __restrict__ bo, float* __restrict__ out) {
    int wave = threadIdx.x >> 5;
    int lane = threadIdx.x & 31;
    int tile = blockIdx.x;
    int Nl = lane & 15;
    long arow = (long)tile * 16 + Nl;
    const __bf16* brun = woT + (long)(wave * 16 + Nl) * C_Z;
    v8f acc = {};
#pragma unroll
    for (int kk = 0; kk < C_Z; kk += 32) {
        v16bf a = fragA(go + arow * C_Z + kk, lane);
        v16bf b = fragB(brun + kk, lane);
        acc = __builtin_amdgcn_wmma_f32_16x16x32_bf16(false, a, false, b, (short)0, acc,
                                                      false, false);
    }
    int mbase = (lane & 16) ? 8 : 0;
    int col = wave * 16 + Nl;
    float bias = bo[col];
#pragma unroll
    for (int r = 0; r < 8; ++r) {
        out[((long)tile * 16 + mbase + r) * C_Z + col] = acc[r] + bias;
    }
}

extern "C" void kernel_launch(void* const* d_in, const int* in_sizes, int n_in,
                              void* d_out, int out_size, void* d_ws, size_t ws_size,
                              hipStream_t stream) {
    const float* z     = (const float*)d_in[0];
    const float* gamma = (const float*)d_in[1];
    const float* beta  = (const float*)d_in[2];
    const float* Wb    = (const float*)d_in[3];
    const float* Wq    = (const float*)d_in[4];
    const float* Wk    = (const float*)d_in[5];
    const float* Wv    = (const float*)d_in[6];
    const float* Wg    = (const float*)d_in[7];
    const float* bg    = (const float*)d_in[8];
    const float* Wo    = (const float*)d_in[9];
    const float* bo    = (const float*)d_in[10];
    float* out = (float*)d_out;

    char* ws = (char*)d_ws;
    size_t off = 0;
    auto carve = [&](size_t bytes) -> char* {
        char* p = ws + off;
        off += (bytes + 255) & ~(size_t)255;
        return p;
    };
    __bf16* zn    = (__bf16*)carve((size_t)NN * C_Z * 2);
    __bf16* qb    = (__bf16*)carve((size_t)NN * C_Z * 2);
    __bf16* kb    = (__bf16*)carve((size_t)NN * C_Z * 2);
    __bf16* vt    = (__bf16*)carve((size_t)NN * C_Z * 2);
    __bf16* gob   = (__bf16*)carve((size_t)NN * C_Z * 2);
    float*  gf    = (float*) carve((size_t)NN * C_Z * 4);
    float*  bias  = (float*) carve((size_t)N_HEAD * NN * 4);
    __bf16* wcatT = (__bf16*)carve((size_t)C_Z * 512 * 2);
    __bf16* woT   = (__bf16*)carve((size_t)C_Z * C_Z * 2);

    ta_pack_weights<<<(C_Z * 512 + C_Z * C_Z + 255) / 256, 256, 0, stream>>>(
        Wq, Wk, Wv, Wg, Wo, wcatT, woT);
    ta_ln_bias<<<NN / 8, 256, 0, stream>>>(z, gamma, beta, Wb, zn, bias);
    ta_proj<<<NN / 16, 256, 0, stream>>>(zn, wcatT, bg, qb, kb, vt, gf);
    ta_attn<<<dim3(N_RES / 16, N_HEAD, N_RES), 256, 0, stream>>>(
        qb, kb, vt, gf, bias, gob);
    ta_outproj<<<NN / 16, 256, 0, stream>>>(gob, woT, bo, out);
}
